// Attention_17042430230543
// MI455X (gfx1250) — compile-verified, matrix operand order checked by reading, not run on hardware
//
#include <hip/hip_runtime.h>

typedef __attribute__((ext_vector_type(16))) _Float16     v16h;
typedef __attribute__((ext_vector_type(8)))  float        v8f;
typedef __attribute__((ext_vector_type(4)))  unsigned int v4u;
typedef __attribute__((ext_vector_type(4)))  int          v4i;
typedef __attribute__((ext_vector_type(8)))  int          v8i_t;

#define EMBED 1024
#define NHEAD 16
#define HDIM  64
#define BATCH 4
#define SEQ   2048
#define MTOT  (BATCH*SEQ)   /* 8192 */
#define BUFH  (128 * 40)    /* one GEMM LDS tile buffer, in halves */

#if defined(__gfx1250__) && __has_builtin(__builtin_amdgcn_tensor_load_to_lds)
#define ATTN_TDM 1
#else
#define ATTN_TDM 0
#endif

// TDM group1 word0 flags: data_size=2B | pad_enable | pad_interval | pad_amount
//  GEMM tile rows are 32 f16 = 16 DW -> pad 4 DW after every 16 DW (stride 40h)
//  K tile rows are 64 f16 = 32 DW    -> pad 4 DW after every 32 DW (stride 72h)
#define TDM_FLAGS_ROW32  0x06D10000
#define TDM_FLAGS_ROW64  0x07110000

union FragU { unsigned int u[8]; uint4 q[2]; v16h v; };

// 16-bit A/B fragment (ISA 7.12.2): per lane, half = lane>>4,
//   u[0..3] = halves [8*half .. 8*half+7],  u[4..7] = halves [16+8*half ..]
// -> two contiguous 16-byte runs: 2 x b128 loads.
__device__ __forceinline__ void load_frag(FragU& f, const _Float16* row, int half) {
  f.q[0] = *(const uint4*)(row + 8 * half);
  f.q[1] = *(const uint4*)(row + 16 + 8 * half);
}

__device__ __forceinline__ v8f zero8() {
  v8f z = {0.f, 0.f, 0.f, 0.f, 0.f, 0.f, 0.f, 0.f};
  return z;
}

#if ATTN_TDM
// 2D f16 tile load via Tensor Data Mover: tile_k x tile_rows from a row-major
// tensor with given row stride (elements), LDS-padded per `flags`.
__device__ __forceinline__ void tdm_load_tile_f16(
    unsigned ldsoff, const _Float16* gsrc,
    int tile_k, int tile_rows, int row_stride, int flags)
{
  unsigned long long ga = (unsigned long long)(const void*)gsrc;
  v4u g0 = {1u, ldsoff, (unsigned)ga,
            ((unsigned)(ga >> 32) & 0x01FFFFFFu) | 0x80000000u};
  v8i_t g1 = {flags,
              tile_k << 16,          // tensor_dim0[15:0] in [31:16]
              tile_rows << 16,       // tensor_dim1[15:0] in [31:16]
              tile_k << 16,          // tile_dim0 in [31:16]
              tile_rows,             // tile_dim1 in [15:0]
              row_stride,            // tensor_dim0_stride[31:0]
              0, 0};
  v4i gz = {0, 0, 0, 0};
#if __clang_major__ >= 23
  v8i_t gz8 = {0, 0, 0, 0, 0, 0, 0, 0};
  __builtin_amdgcn_tensor_load_to_lds(g0, g1, gz, gz, gz8, 0);
#else
  __builtin_amdgcn_tensor_load_to_lds(g0, g1, gz, gz, 0);
#endif
}

__device__ __forceinline__ void tdm_wait() {
#if __has_builtin(__builtin_amdgcn_s_wait_tensorcnt)
  __builtin_amdgcn_s_wait_tensorcnt(0);
#else
  asm volatile("s_wait_tensorcnt 0x0" ::: "memory");
#endif
}
#endif

// ---------------------------------------------------------------------------
// f32 -> f16 pre-conversion (one pass; removes all cvt work from GEMM loops)
// ---------------------------------------------------------------------------
__global__ __launch_bounds__(256) void cvt_f32_f16(
    const float* __restrict__ s, _Float16* __restrict__ d, int n4)
{
  int i = blockIdx.x * 256 + threadIdx.x;
  if (i < n4) {
    float4 f = ((const float4*)s)[i];
    union { _Float16 h[4]; uint2 u; } pk;
    pk.h[0] = (_Float16)f.x; pk.h[1] = (_Float16)f.y;
    pk.h[2] = (_Float16)f.z; pk.h[3] = (_Float16)f.w;
    ((uint2*)d)[i] = pk.u;
  }
}

// ---------------------------------------------------------------------------
// Shared GEMM main loop: C[128x128] tile at (m0,n0) of A[.,1024] x B[.,1024]^T,
// both f16 row-major with K contiguous. Double-buffered LDS; tiles streamed by
// the TDM while WMMAs consume the other buffer.
// ---------------------------------------------------------------------------
__device__ __forceinline__ void gemm_mainloop(
    const _Float16* __restrict__ A, const _Float16* __restrict__ B,
    _Float16* As, _Float16* Bs, int m0, int n0, int tid, v8f (&acc)[2][4])
{
  const int lane = tid & 31, wave = tid >> 5;
  const int half = lane >> 4, lanelo = lane & 15;
  const int wm = wave & 3, wn = wave >> 2;
  int cur = 0;

#if ATTN_TDM
  if (wave == 0) {
    tdm_load_tile_f16((unsigned)(unsigned long long)(void*)&As[0],
                      A + (size_t)m0 * 1024, 32, 128, 1024, TDM_FLAGS_ROW32);
    tdm_load_tile_f16((unsigned)(unsigned long long)(void*)&Bs[0],
                      B + (size_t)n0 * 1024, 32, 128, 1024, TDM_FLAGS_ROW32);
    tdm_wait();
  }
  __syncthreads();
  for (int k0 = 0; k0 < 1024; k0 += 32) {
    const bool hasnext = (k0 + 32 < 1024);
    const int nxt = cur ^ 1;
    if (hasnext && wave == 0) {
      tdm_load_tile_f16((unsigned)(unsigned long long)(void*)&As[nxt * BUFH],
                        A + (size_t)m0 * 1024 + k0 + 32, 32, 128, 1024,
                        TDM_FLAGS_ROW32);
      tdm_load_tile_f16((unsigned)(unsigned long long)(void*)&Bs[nxt * BUFH],
                        B + (size_t)n0 * 1024 + k0 + 32, 32, 128, 1024,
                        TDM_FLAGS_ROW32);
    }
    FragU a[2], b[4];
    for (int mi = 0; mi < 2; ++mi)
      load_frag(a[mi], &As[cur * BUFH + (wm * 32 + mi * 16 + lanelo) * 40], half);
    for (int ni = 0; ni < 4; ++ni)
      load_frag(b[ni], &Bs[cur * BUFH + (wn * 64 + ni * 16 + lanelo) * 40], half);
    for (int mi = 0; mi < 2; ++mi)
      for (int ni = 0; ni < 4; ++ni)
        acc[mi][ni] = __builtin_amdgcn_wmma_f32_16x16x32_f16(
            false, a[mi].v, false, b[ni].v, (short)0, acc[mi][ni], false, false);
    if (hasnext) {
      if (wave == 0) tdm_wait();
      __syncthreads();
      cur = nxt;
    }
  }
#else
  // Fallback: manual single-buffer staging (also used for host-side parse).
  for (int k0 = 0; k0 < 1024; k0 += 32) {
    __syncthreads();
    for (int p = 0; p < 2; ++p) {
      int ch = tid + 256 * p;             // 512 16B-chunks per 128x32 tile
      int row = ch >> 2, off = (ch & 3) * 8;
      *(uint4*)&As[row * 40 + off] =
          *(const uint4*)(A + (size_t)(m0 + row) * 1024 + k0 + off);
      *(uint4*)&Bs[row * 40 + off] =
          *(const uint4*)(B + (size_t)(n0 + row) * 1024 + k0 + off);
    }
    __syncthreads();
    FragU a[2], b[4];
    for (int mi = 0; mi < 2; ++mi)
      load_frag(a[mi], &As[(wm * 32 + mi * 16 + lanelo) * 40], half);
    for (int ni = 0; ni < 4; ++ni)
      load_frag(b[ni], &Bs[(wn * 64 + ni * 16 + lanelo) * 40], half);
    for (int mi = 0; mi < 2; ++mi)
      for (int ni = 0; ni < 4; ++ni)
        acc[mi][ni] = __builtin_amdgcn_wmma_f32_16x16x32_f16(
            false, a[mi].v, false, b[ni].v, (short)0, acc[mi][ni], false, false);
  }
  (void)cur;
#endif
}

// ---------------------------------------------------------------------------
// Kernel 1: QKV projection (f16 x f16, f32 accum) -> Q/K/V f16 in [B,H,T,D].
// ---------------------------------------------------------------------------
__global__ __launch_bounds__(256) void qkv_gemm(
    const _Float16* __restrict__ Xh, const _Float16* __restrict__ Wh,
    const float* __restrict__ bias,
    _Float16* __restrict__ Qo, _Float16* __restrict__ Ko, _Float16* __restrict__ Vo)
{
  __shared__ __align__(16) _Float16 As[2 * BUFH];
  __shared__ __align__(16) _Float16 Bs[2 * BUFH];
  const int tid = threadIdx.x;
  const int n0 = blockIdx.x * 128;
  const int m0 = blockIdx.y * 128;
  const int lane = tid & 31, wave = tid >> 5;
  const int half = lane >> 4, lanelo = lane & 15;
  const int wm = wave & 3, wn = wave >> 2;

  v8f acc[2][4];
  for (int mi = 0; mi < 2; ++mi)
    for (int ni = 0; ni < 4; ++ni) acc[mi][ni] = zero8();

  gemm_mainloop(Xh, Wh, As, Bs, m0, n0, tid, acc);

  for (int mi = 0; mi < 2; ++mi)
    for (int ni = 0; ni < 4; ++ni) {
      int n = n0 + wn * 64 + ni * 16 + lanelo;      // 0..3071
      int seg = n >> 10, c = n & 1023;
      int h = c >> 6, d = c & 63;
      _Float16* dst = (seg == 0) ? Qo : ((seg == 1) ? Ko : Vo);
      float bv = bias[n];
      for (int r = 0; r < 8; ++r) {
        int m = m0 + wm * 32 + mi * 16 + r + 8 * half;
        int bb = m >> 11, t = m & 2047;
        dst[((size_t)(bb * NHEAD + h) * SEQ + t) * HDIM + d] =
            (_Float16)(acc[mi][ni][r] + bv);
      }
    }
}

// ---------------------------------------------------------------------------
// Kernel 2: causal flash attention. K tile via TDM (hardware-padded), V tile
// transposed manually, P round-trips LDS to reach A-fragment layout.
// ---------------------------------------------------------------------------
__global__ __launch_bounds__(256) void attn_kernel(
    const _Float16* __restrict__ Qg, const _Float16* __restrict__ Kg,
    const _Float16* __restrict__ Vg, const int* __restrict__ amask,
    _Float16* __restrict__ Og)
{
  __shared__ __align__(16) _Float16 Ks[64 * 72];       // [key][d]
  __shared__ __align__(16) _Float16 Vt[64 * 72];       // [d][key]
  __shared__ __align__(16) _Float16 Ps[8 * 16 * 72];   // per-wave P staging
  const int tid = threadIdx.x;
  const int lane = tid & 31, wave = tid >> 5;
  const int half = lane >> 4, lanelo = lane & 15;
  const int qb = blockIdx.x, h = blockIdx.y, b = blockIdx.z;
  const size_t bh = (size_t)(b * NHEAD + h);

  const int tq_lane = qb * 128 + wave * 16 + lanelo;
  const _Float16* qrow = Qg + (bh * SEQ + tq_lane) * HDIM;
  FragU qf[2];
  load_frag(qf[0], qrow, half);
  load_frag(qf[1], qrow + 32, half);

  int tqr[8], amq[8];
  for (int r = 0; r < 8; ++r) {
    tqr[r] = qb * 128 + wave * 16 + r + 8 * half;
    amq[r] = amask[b * SEQ + tqr[r]];
  }

  v8f o[4];
  float mrow[8], lrow[8];
  for (int ni = 0; ni < 4; ++ni) o[ni] = zero8();
  for (int r = 0; r < 8; ++r) { mrow[r] = -1e30f; lrow[r] = 0.f; }

  const int kblocks = 2 * (qb + 1);   // uniform per workgroup (causal limit)
  const float scale = 0.125f;         // 1/sqrt(64)
  const int pbase = wave * 16 * 72;

  for (int kb = 0; kb < kblocks; ++kb) {
    const int tk0 = kb * 64;
    __syncthreads();

#if ATTN_TDM
    if (wave == 0)
      tdm_load_tile_f16((unsigned)(unsigned long long)(const void*)Ks,
                        Kg + (bh * SEQ + tk0) * HDIM, 64, 64, 64,
                        TDM_FLAGS_ROW64);
    {
      int rowt = tid >> 3;          // 0..31
      int cg = (tid & 7) * 8;       // 0..56
      for (int p = 0; p < 2; ++p) {
        int tkl = rowt + 32 * p;
        union { uint4 u; _Float16 e[8]; } vv;
        vv.u = *(const uint4*)(Vg + (bh * SEQ + tk0 + tkl) * HDIM + cg);
        for (int e = 0; e < 8; ++e) Vt[(cg + e) * 72 + tkl] = vv.e[e];
      }
    }
    __syncthreads();
    if (wave == 0) tdm_wait();
    __syncthreads();
#else
    {
      int rowt = tid >> 3;
      int cg = (tid & 7) * 8;
      for (int p = 0; p < 2; ++p) {
        int tkl = rowt + 32 * p;
        uint4 kv = *(const uint4*)(Kg + (bh * SEQ + tk0 + tkl) * HDIM + cg);
        *(uint4*)&Ks[tkl * 72 + cg] = kv;
        union { uint4 u; _Float16 e[8]; } vv;
        vv.u = *(const uint4*)(Vg + (bh * SEQ + tk0 + tkl) * HDIM + cg);
        for (int e = 0; e < 8; ++e) Vt[(cg + e) * 72 + tkl] = vv.e[e];
      }
    }
    __syncthreads();
#endif

    // S = Q K^T (16 x 64 per wave)
    v8f s[4];
    for (int ni = 0; ni < 4; ++ni) s[ni] = zero8();
    for (int ks = 0; ks < 2; ++ks) {
      FragU bf[4];
      for (int ni = 0; ni < 4; ++ni)
        load_frag(bf[ni], &Ks[(ni * 16 + lanelo) * 72 + 32 * ks], half);
      for (int ni = 0; ni < 4; ++ni)
        s[ni] = __builtin_amdgcn_wmma_f32_16x16x32_f16(
            false, qf[ks].v, false, bf[ni].v, (short)0, s[ni], false, false);
    }

    // mask + online softmax
    int amk[4];
    for (int ni = 0; ni < 4; ++ni)
      amk[ni] = amask[b * SEQ + tk0 + ni * 16 + lanelo];
    float sv[4][8], rmax[8];
    for (int r = 0; r < 8; ++r) rmax[r] = -1e30f;
    for (int ni = 0; ni < 4; ++ni) {
      int tk = tk0 + ni * 16 + lanelo;
      for (int r = 0; r < 8; ++r) {
        float x = s[ni][r] * scale;
        bool ok = (tk <= tqr[r]) && (amk[ni] != 0) && (amq[r] != 0);
        x = ok ? x : -1e30f;
        sv[ni][r] = x;
        rmax[r] = fmaxf(rmax[r], x);
      }
    }
    for (int r = 0; r < 8; ++r) {
      float v = rmax[r];
      for (int off = 1; off < 16; off <<= 1)
        v = fmaxf(v, __shfl_xor(v, off, 32));
      rmax[r] = v;
    }
    float alpha[8];
    for (int r = 0; r < 8; ++r) {
      float mn = fmaxf(mrow[r], rmax[r]);
      alpha[r] = __expf(mrow[r] - mn);
      mrow[r] = mn;
      lrow[r] *= alpha[r];
    }
    for (int ni = 0; ni < 4; ++ni)
      for (int r = 0; r < 8; ++r) o[ni][r] *= alpha[r];

    float rsum[8];
    for (int r = 0; r < 8; ++r) rsum[r] = 0.f;
    for (int ni = 0; ni < 4; ++ni)
      for (int r = 0; r < 8; ++r) {
        float p = __expf(sv[ni][r] - mrow[r]);
        rsum[r] += p;
        Ps[pbase + (r + 8 * half) * 72 + ni * 16 + lanelo] = (_Float16)p;
      }
    for (int r = 0; r < 8; ++r) {
      float v = rsum[r];
      for (int off = 1; off < 16; off <<= 1) v += __shfl_xor(v, off, 32);
      lrow[r] += v;
    }
    __syncthreads();   // P visible, Ks reads done

    // O += P * V
    for (int ks = 0; ks < 2; ++ks) {
      FragU pf;
      load_frag(pf, &Ps[pbase + lanelo * 72 + 32 * ks], half);
      for (int ni = 0; ni < 4; ++ni) {
        FragU vf;
        load_frag(vf, &Vt[(ni * 16 + lanelo) * 72 + 32 * ks], half);
        o[ni] = __builtin_amdgcn_wmma_f32_16x16x32_f16(
            false, pf.v, false, vf.v, (short)0, o[ni], false, false);
      }
    }
  }

  for (int r = 0; r < 8; ++r) {
    float inv = (lrow[r] > 0.f) ? (1.f / lrow[r]) : 0.f;
    for (int ni = 0; ni < 4; ++ni)
      Og[(size_t)(b * SEQ + tqr[r]) * EMBED + h * HDIM + ni * 16 + lanelo] =
          (_Float16)(o[ni][r] * inv);
  }
}

// ---------------------------------------------------------------------------
// Kernel 3: output projection (f16 x f16) + bias -> f32 out.
// ---------------------------------------------------------------------------
__global__ __launch_bounds__(256) void out_gemm(
    const _Float16* __restrict__ A, const _Float16* __restrict__ Wh,
    const float* __restrict__ bias, float* __restrict__ out)
{
  __shared__ __align__(16) _Float16 As[2 * BUFH];
  __shared__ __align__(16) _Float16 Bs[2 * BUFH];
  const int tid = threadIdx.x;
  const int n0 = blockIdx.x * 128;
  const int m0 = blockIdx.y * 128;
  const int lane = tid & 31, wave = tid >> 5;
  const int half = lane >> 4, lanelo = lane & 15;
  const int wm = wave & 3, wn = wave >> 2;

  v8f acc[2][4];
  for (int mi = 0; mi < 2; ++mi)
    for (int ni = 0; ni < 4; ++ni) acc[mi][ni] = zero8();

  gemm_mainloop(A, Wh, As, Bs, m0, n0, tid, acc);

  for (int mi = 0; mi < 2; ++mi)
    for (int ni = 0; ni < 4; ++ni) {
      int n = n0 + wn * 64 + ni * 16 + lanelo;
      float bv = bias[n];
      for (int r = 0; r < 8; ++r) {
        int m = m0 + wm * 32 + mi * 16 + r + 8 * half;
        out[(size_t)m * 1024 + n] = acc[mi][ni][r] + bv;
      }
    }
}

// ---------------------------------------------------------------------------
extern "C" void kernel_launch(void* const* d_in, const int* in_sizes, int n_in,
                              void* d_out, int out_size, void* d_ws, size_t ws_size,
                              hipStream_t stream) {
  const float* X     = (const float*)d_in[0];   // [B,T,C]
  const int*   amask = (const int*)d_in[1];     // [B,T]
  const float* Wqkv  = (const float*)d_in[2];   // [3C,C]
  const float* bqkv  = (const float*)d_in[3];   // [3C]
  const float* Wo    = (const float*)d_in[4];   // [C,C]
  const float* bo    = (const float*)d_in[5];   // [C]
  float* out = (float*)d_out;

  const size_t elems = (size_t)MTOT * EMBED;    // 8,388,608
  _Float16* Q     = (_Float16*)d_ws;
  _Float16* K     = Q + elems;
  _Float16* V     = K + elems;
  _Float16* Ao    = V + elems;
  _Float16* Xh    = Ao + elems;
  _Float16* Wqh   = Xh + elems;                 // 3,145,728 elems
  _Float16* Woh   = Wqh + (size_t)3 * EMBED * EMBED;

  cvt_f32_f16<<<dim3((int)(elems / 4 / 256)), 256, 0, stream>>>(X, Xh, (int)(elems / 4));
  cvt_f32_f16<<<dim3(3 * EMBED * EMBED / 4 / 256), 256, 0, stream>>>(
      Wqkv, Wqh, 3 * EMBED * EMBED / 4);
  cvt_f32_f16<<<dim3(EMBED * EMBED / 4 / 256), 256, 0, stream>>>(
      Wo, Woh, EMBED * EMBED / 4);

  qkv_gemm<<<dim3(24, 64), 256, 0, stream>>>(Xh, Wqh, bqkv, Q, K, V);
  attn_kernel<<<dim3(SEQ / 128, NHEAD, BATCH), 256, 0, stream>>>(Q, K, V, amask, Ao);
  out_gemm<<<dim3(8, 64), 256, 0, stream>>>(Ao, Woh, bo, out);
}
